// W2V_SkipGram_45088566673655
// MI455X (gfx1250) — matrix operand order repaired and checked by reading, not compiled, not run on hardware
//
#include <hip/hip_runtime.h>

// ---- problem constants (match reference) ----
#define NDIM   128
#define NS     10
#define BATCH  16384
#define WM1    4          // WINDOW-1 context rows
#define NROWS  44         // 4 context + 40 negatives per batch element
#define WAVES_PER_BLOCK 8
#define LOG1EM9 (-20.7232658f)   // log(1e-9), reference's sigmoid-underflow patch

typedef __attribute__((ext_vector_type(16))) _Float16 v16h;
typedef __attribute__((ext_vector_type(8)))  float    v8f;

__device__ __forceinline__ v16h pack16(float4 a, float4 b, float4 c, float4 d) {
    v16h r;
    r[0]=(_Float16)a.x;  r[1]=(_Float16)a.y;  r[2]=(_Float16)a.z;  r[3]=(_Float16)a.w;
    r[4]=(_Float16)b.x;  r[5]=(_Float16)b.y;  r[6]=(_Float16)b.z;  r[7]=(_Float16)b.w;
    r[8]=(_Float16)c.x;  r[9]=(_Float16)c.y;  r[10]=(_Float16)c.z; r[11]=(_Float16)c.w;
    r[12]=(_Float16)d.x; r[13]=(_Float16)d.y; r[14]=(_Float16)d.z; r[15]=(_Float16)d.w;
    return r;
}

// fast, stable log(sigmoid(x)) = min(x,0) - log(1 + exp(-|x|))
// log argument is in (1,2] -> single v_exp_f32 + v_log_f32, ample accuracy for a mean loss
__device__ __forceinline__ float log_sigmoid_fast(float x) {
    return fminf(x, 0.0f) - __logf(1.0f + __expf(-fabsf(x)));
}

// One wave32 per batch element.
// Score rows: 0..3 = context (pos), 4..43 = negatives (w-major, s-minor), 44..47 = pad.
__global__ __launch_bounds__(32 * WAVES_PER_BLOCK)
void w2v_skipgram_wmma_kernel(const int*   __restrict__ input_ids, // [5, BATCH]
                              const int*   __restrict__ nsi,       // [4, BATCH, NS]
                              const float* __restrict__ W,         // [NVOCAB, NDIM]
                              double*      __restrict__ ws)        // [2]: pos_sum16, neg_sum16
{
    const int lane = threadIdx.x & 31;
    const int b    = blockIdx.x * WAVES_PER_BLOCK + (threadIdx.x >> 5);

    const int  mlane  = lane & 15;        // A layout: row M = lane & 15
    const bool hi     = lane >= 16;
    const int  kbaseA = hi ? 8  : 0;      // A: lane<16 -> K {0..7,16..23}, lane>=16 -> K {8..15,24..31}
    const int  kbaseB = hi ? 8  : 0;      // B: lane<16 -> K {0..7,16..23}, lane>=16 -> K {8..15,24..31}

    // ---- target embedding vi, broadcast into all 16 B columns ----
    const int    vi_id = input_ids[2 * BATCH + b];
    const float* vi    = W + (size_t)vi_id * NDIM;

    v16h bfrag[4];
    #pragma unroll
    for (int c = 0; c < 4; ++c) {
        const float* p = vi + c * 32 + kbaseB;
        bfrag[c] = pack16(*(const float4*)(p + 0),  *(const float4*)(p + 4),
                          *(const float4*)(p + 16), *(const float4*)(p + 20));
    }

    float pos_sum = 0.0f, neg_sum = 0.0f;

    #pragma unroll
    for (int t = 0; t < 3; ++t) {
        // row id this lane-pair gathers for A (pad rows reuse row 0; masked at accumulation)
        const int m = t * 16 + mlane;
        int id = 0;
        if (m < WM1) {
            const int wrow = (m < 2) ? m : m + 1;          // skip center row (TID=2)
            id = input_ids[wrow * BATCH + b];
        } else if (m < NROWS) {
            const int j = m - WM1;
            const int w = j / NS;
            const int s = j - w * NS;
            id = nsi[(w * BATCH + b) * NS + s];
        }
        const float* rowp = W + (size_t)id * NDIM;

        v8f acc = {0.f, 0.f, 0.f, 0.f, 0.f, 0.f, 0.f, 0.f};
        #pragma unroll
        for (int c = 0; c < 4; ++c) {
            const float* p = rowp + c * 32 + kbaseA;
            v16h afrag = pack16(*(const float4*)(p + 0),  *(const float4*)(p + 4),
                                *(const float4*)(p + 16), *(const float4*)(p + 20));
            // D = A(16x32 f16) * B(32x16 f16) + C(16x16 f32)
            acc = __builtin_amdgcn_wmma_f32_16x16x32_f16(
                      false, afrag, false, bfrag[c], (short)0, acc, false, false);
        }

        // D layout: VGPR r holds row M = r + 8*hi; all 16 N-columns are duplicates.
        // Branch-free accumulation (selects lower to v_cndmask, no EXEC divergence).
        #pragma unroll
        for (int r = 0; r < 8; ++r) {
            const float sc = acc[r];
            if (t == 0 && r < 4) {
                // lo half: pos row r      -> log sigmoid(sc)
                // hi half: neg row r + 8  -> log sigmoid(-sc) with underflow patch
                const float x  = hi ? -sc : sc;
                const float ls = log_sigmoid_fast(x);
                const float ln = (sc > 103.0f) ? LOG1EM9 : ls;
                pos_sum += hi ? 0.0f : ls;
                neg_sum += hi ? ln   : 0.0f;
            } else if (t == 2 && r >= 4) {
                // lo half: neg rows 36..39 ; hi half: pad rows 44..47 (masked out)
                float ls = log_sigmoid_fast(-sc);
                ls = (sc > 103.0f) ? LOG1EM9 : ls;
                neg_sum += hi ? 0.0f : ls;
            } else {
                // both halves are valid neg rows
                float ls = log_sigmoid_fast(-sc);
                ls = (sc > 103.0f) ? LOG1EM9 : ls;
                neg_sum += ls;
            }
        }
    }

    // wave reduction (every score duplicated 16x across N-columns -> /16 at finalize)
    #pragma unroll
    for (int off = 16; off > 0; off >>= 1) {
        pos_sum += __shfl_xor(pos_sum, off, 32);
        neg_sum += __shfl_xor(neg_sum, off, 32);
    }
    if (lane == 0) {
        atomicAdd(&ws[0], (double)pos_sum);
        atomicAdd(&ws[1], (double)neg_sum);
    }
}

__global__ void w2v_finalize_kernel(const double* __restrict__ ws, float* __restrict__ out) {
    const double pos_mean = ws[0] / (16.0 * (double)WM1 * (double)BATCH);
    const double neg_mean = ws[1] / (16.0 * (double)(WM1 * NS) * (double)BATCH);
    out[0] = (float)(-(pos_mean + neg_mean));
}

extern "C" void kernel_launch(void* const* d_in, const int* in_sizes, int n_in,
                              void* d_out, int out_size, void* d_ws, size_t ws_size,
                              hipStream_t stream) {
    (void)in_sizes; (void)n_in; (void)out_size; (void)ws_size;
    const int*   input_ids = (const int*)d_in[0];   // [5, BATCH]
    const int*   nsi       = (const int*)d_in[1];   // [4, BATCH, NS]
    const float* W         = (const float*)d_in[2]; // [NVOCAB, NDIM]
    float*  out = (float*)d_out;
    double* ws  = (double*)d_ws;

    hipMemsetAsync(d_ws, 0, 2 * sizeof(double), stream);

    dim3 block(32 * WAVES_PER_BLOCK);
    dim3 grid(BATCH / WAVES_PER_BLOCK);
    w2v_skipgram_wmma_kernel<<<grid, block, 0, stream>>>(input_ids, nsi, W, ws);
    w2v_finalize_kernel<<<1, 1, 0, stream>>>(ws, out);
}
